// LVShapeNet_47966194762188
// MI455X (gfx1250) — matrix operand order repaired and checked by reading, not compiled, not run on hardware
//
#include <hip/hip_runtime.h>
#include <math.h>

// ---------------------------------------------------------------------------
// LVShapeNet forward for MI455X (gfx1250, wave32, WMMA).
// Convolutions are implicit-GEMM on v_wmma_f32_16x16x32_f16 (f32 accum),
// activations/weights carried in f16 (BN renormalizes each layer so fp16
// staging is numerically benign). Voxelization is VALU/exp bound by nature.
// ---------------------------------------------------------------------------

typedef __attribute__((ext_vector_type(16))) _Float16 v16h;
typedef __attribute__((ext_vector_type(8)))  float    v8f;

#define B_    8
#define VOXD  32
#define S_    12
#define NPTS_ 32
#define NTET  1056          // (S-1)*NPTS*3
#define KSH   20.0f

__device__ __forceinline__ float sigm(float x) { return 1.0f / (1.0f + __expf(-x)); }

// ---------------------------------------------------------------------------
// 1) Per-batch min/max normalization, f32 -> f16
// ---------------------------------------------------------------------------
__global__ __launch_bounds__(256) void norm_input(const float* __restrict__ img,
                                                  _Float16* __restrict__ y) {
  const int b = blockIdx.x;
  const float* p = img + (size_t)b * 32768;
  float mn = 1e30f, mx = -1e30f;
  for (int i = threadIdx.x; i < 32768; i += 256) {
    float v = p[i];
    mn = fminf(mn, v);
    mx = fmaxf(mx, v);
  }
  __shared__ float smn[256], smx[256];
  smn[threadIdx.x] = mn; smx[threadIdx.x] = mx;
  __syncthreads();
  for (int o = 128; o > 0; o >>= 1) {
    if (threadIdx.x < o) {
      smn[threadIdx.x] = fminf(smn[threadIdx.x], smn[threadIdx.x + o]);
      smx[threadIdx.x] = fmaxf(smx[threadIdx.x], smx[threadIdx.x + o]);
    }
    __syncthreads();
  }
  const float MN = smn[0];
  const float inv = 1.0f / (smx[0] - MN + 0.01f);
  for (int i = threadIdx.x; i < 32768; i += 256)
    y[(size_t)b * 32768 + i] = (_Float16)((p[i] - MN) * inv);
}

// ---------------------------------------------------------------------------
// 2) Implicit-GEMM conv3d (ks^3, pad) via WMMA f16 -> f32.
//    GEMM view: out[co, n] = sum_k W[co,k] * patch[k,n],
//      n = ((b*D+z)*D+y)*D+x over N = 8*D^3,  k = ((ci*ks+dz)*ks+dy)*ks+dx.
//    Workgroup: 256 threads = 8 wave32.  WG tile = 32 M x 128 N:
//      wave w -> M-tile (w&1), N-group (w>>1)*32; each wave owns two 16-wide
//      N-subtiles -> 2 WMMA per K-step reusing one A fragment.
//    Gathers are BRANCHLESS (clamped index + select) so the unrolled loads
//    issue as a clause with grouped waits (memory-level parallelism), and the
//    aligned A path is a single b128 load.
// ---------------------------------------------------------------------------
__global__ __launch_bounds__(256) void conv3d_wmma(
    const _Float16* __restrict__ act,   // [8, Cin, D^3] f16
    const float*    __restrict__ wgt,   // [Cout, Cin*ks^3] f32
    const float*    __restrict__ bias,  // [Cout]
    _Float16*       __restrict__ out,   // [8, Cout, D^3] f16 (pre-BN, +bias)
    int Cin, int Cout, int D, int ks, int pad) {
  const int ks2  = ks * ks;
  const int kvol = ks2 * ks;
  const int K    = Cin * kvol;
  const int D2   = D * D;
  const int D3   = D2 * D;
  const int Ntot = B_ * D3;
  const int nbase = blockIdx.x * 128;
  const int mbase = blockIdx.y * 32;
  const int tid  = threadIdx.x;
  const int lane = tid & 31;          // wave32 lane
  const int wv   = tid >> 5;          // wave id 0..7
  const int mloc = (wv & 1) * 16;     // M sub-tile within 32
  const int nloc = (wv >> 1) * 32;    // N group (two 16-wide subtiles)

  __shared__ _Float16 As[32 * 32];    // [mlocal][k], k contiguous
  __shared__ _Float16 Bs[128 * 32];   // [nlocal][k], k contiguous

  v8f acc0 = {}, acc1 = {};

  // A staging map: le = tid*4+q  -> row aml = tid>>3, k = (tid&7)*4+q (contig)
  const int aml = tid >> 3;
  // B staging map: le = tid*16+q -> col bnl = tid>>1, k = (tid&1)*16+q
  const int bnl  = tid >> 1;
  const int bkl0 = (tid & 1) * 16;
  const int bn   = nbase + bnl;
  const bool nvalid = bn < Ntot;
  int bb = 0, bz = 0, by = 0, bx = 0;
  if (nvalid) {
    bb = bn / D3;
    const int rz = bn - bb * D3;
    bz = rz / D2; by = (rz / D) % D; bx = rz % D;
  }
  const bool k_aligned = (K & 3) == 0;

  const int ksteps = (K + 31) >> 5;
  for (int s = 0; s < ksteps; ++s) {
    const int k0 = s << 5;

    // Warm L2 with next weight tile (global_prefetch_b8).
    if (s + 1 < ksteps && tid < 32) {
      const float* pf = wgt + (size_t)(mbase + tid) * K + (k0 + 32);
      __builtin_prefetch(pf, 0, 1);
    }

    // ---- gather A into regs: aligned fast path = one b128 load ----
    float av[4];
    const int arow = mbase + aml;
    const int akbase = k0 + (tid & 7) * 4;
    if (k_aligned && (akbase + 3 < K)) {
      const float4 f4 = *(const float4*)(wgt + (size_t)arow * K + akbase);
      av[0] = f4.x; av[1] = f4.y; av[2] = f4.z; av[3] = f4.w;
    } else {
#pragma unroll
      for (int q = 0; q < 4; ++q) {
        const int kk = akbase + q;
        const bool inb = kk < K;
        const float v = wgt[(size_t)arow * K + (inb ? kk : 0)];
        av[q] = inb ? v : 0.0f;
      }
    }
    // ---- gather B into regs (branchless: clamped index + select) ----
    _Float16 bv[16];
    {
      const int kk0 = k0 + bkl0;
      int ci = kk0 / kvol;
      int r  = kk0 - ci * kvol;
      int dz = r / ks2;
      const int rem = r - dz * ks2;
      int dy = rem / ks;
      int dx = rem - dy * ks;
#pragma unroll
      for (int q = 0; q < 16; ++q) {
        const int iz = bz + dz - pad, iy = by + dy - pad, ix = bx + dx - pad;
        const bool inb = nvalid && (kk0 + q) < K &&
                         (unsigned)iz < (unsigned)D && (unsigned)iy < (unsigned)D &&
                         (unsigned)ix < (unsigned)D;
        const size_t idx =
            inb ? (((size_t)bb * Cin + ci) * D3 + (size_t)iz * D2 +
                   (size_t)(iy * D + ix))
                : 0;
        const _Float16 v = act[idx];          // unconditional load
        bv[q] = inb ? v : (_Float16)0.0f;     // zero-fill via select
        // carry-increment (dx,dy,dz,ci) for next k
        if (++dx == ks) {
          dx = 0;
          if (++dy == ks) {
            dy = 0;
            if (++dz == ks) { dz = 0; ++ci; }
          }
        }
      }
    }

    __syncthreads();  // prior step's fragment reads complete
#pragma unroll
    for (int q = 0; q < 4; ++q) As[aml * 32 + (tid & 7) * 4 + q] = (_Float16)av[q];
#pragma unroll
    for (int q = 0; q < 16; ++q) Bs[bnl * 32 + bkl0 + q] = bv[q];
    __syncthreads();

    // ---- fragments, A and B both in contiguous k = 16*half + s order (wave32, EXEC all-1s) ----
    const int half = lane >> 4;
    const int mrow = mloc + (lane & 15);
    v16h a;
    // A 16x32: lanes<16 hold K 0..15, lanes>=16 hold K 16..31
#pragma unroll
    for (int i = 0; i < 4; ++i) {
      const int k = half * 16 + 2 * i;
      a[2 * i]     = As[mrow * 32 + k];
      a[2 * i + 1] = As[mrow * 32 + k + 1];
    }
#pragma unroll
    for (int i = 4; i < 8; ++i) {
      const int k = half * 16 + 8 + 2 * (i - 4);
      a[2 * i]     = As[mrow * 32 + k];
      a[2 * i + 1] = As[mrow * 32 + k + 1];
    }
    // B 32x16: lanes<16 hold K 0..15, lanes>=16 hold K 16..31 (2 per VGPR)
    v16h b0, b1;
    const int nc0 = nloc + (lane & 15);
    const int nc1 = nc0 + 16;
#pragma unroll
    for (int i = 0; i < 8; ++i) {
      const int k = half * 16 + 2 * i;
      b0[2 * i]     = Bs[nc0 * 32 + k];
      b0[2 * i + 1] = Bs[nc0 * 32 + k + 1];
      b1[2 * i]     = Bs[nc1 * 32 + k];
      b1[2 * i + 1] = Bs[nc1 * 32 + k + 1];
    }
    acc0 = __builtin_amdgcn_wmma_f32_16x16x32_f16(
        false, a, false, b0, (short)0, acc0, false, false);
    acc1 = __builtin_amdgcn_wmma_f32_16x16x32_f16(
        false, a, false, b1, (short)0, acc1, false, false);
  }

  // Store C/D fragments: lane l holds N = l&15, M = r + 8*(l>>4) per VGPR r.
  const int mrow0 = mbase + mloc + (lane >> 4) * 8;
#pragma unroll
  for (int hn = 0; hn < 2; ++hn) {
    const int ncolg = nbase + nloc + hn * 16 + (lane & 15);
    if (ncolg >= Ntot) continue;
    const int bb2 = ncolg / D3;
    const int rz = ncolg - bb2 * D3;
    const int z = rz / D2, yy = (rz / D) % D, xx = rz % D;
    const v8f accs = hn ? acc1 : acc0;
#pragma unroll
    for (int r = 0; r < 8; ++r) {
      const int co = mrow0 + r;
      if (co < Cout)
        out[((size_t)bb2 * Cout + co) * D3 + (size_t)z * D2 + yy * D + xx] =
            (_Float16)(accs[r] + bias[co]);
    }
  }
}

// ---------------------------------------------------------------------------
// 3) BatchNorm statistics (biased var over batch+spatial), one block/channel.
// ---------------------------------------------------------------------------
__global__ __launch_bounds__(256) void bn_stats(const _Float16* __restrict__ x,
                                                float* __restrict__ mean,
                                                float* __restrict__ var,
                                                int C, int D3) {
  const int c = blockIdx.x;
  const int tid = threadIdx.x;
  float s = 0.0f, s2 = 0.0f;
  for (int b = 0; b < B_; ++b) {
    const _Float16* p = x + ((size_t)b * C + c) * D3;
    for (int i = tid; i < D3; i += 256) {
      const float v = (float)p[i];
      s += v;
      s2 += v * v;
    }
  }
  __shared__ float ss[256], ss2[256];
  ss[tid] = s; ss2[tid] = s2;
  __syncthreads();
  for (int o = 128; o > 0; o >>= 1) {
    if (tid < o) { ss[tid] += ss[tid + o]; ss2[tid] += ss2[tid + o]; }
    __syncthreads();
  }
  if (tid == 0) {
    const float n = (float)(B_ * D3);
    const float m = ss[0] / n;
    mean[c] = m;
    var[c] = ss2[0] / n - m * m;
  }
}

// ---------------------------------------------------------------------------
// 4) BN apply + ReLU + optional 2x2x2 maxpool; f16 out (+optional f32 copy).
// ---------------------------------------------------------------------------
__global__ __launch_bounds__(256) void bn_relu_pool(
    const _Float16* __restrict__ x, const float* __restrict__ mean,
    const float* __restrict__ var, const float* __restrict__ g,
    const float* __restrict__ beta, _Float16* __restrict__ y,
    float* __restrict__ y32, int C, int Din, int pool) {
  const int Dout = pool ? (Din >> 1) : Din;
  const int Do3 = Dout * Dout * Dout;
  const size_t total = (size_t)B_ * C * Do3;
  const size_t idx = (size_t)blockIdx.x * 256 + threadIdx.x;
  if (idx >= total) return;
  const int r = (int)(idx % Do3);
  const int c = (int)((idx / Do3) % C);
  const int b = (int)(idx / ((size_t)Do3 * C));
  const int z = r / (Dout * Dout), yy = (r / Dout) % Dout, xx = r % Dout;
  const float sc = g[c] * rsqrtf(var[c] + 1e-5f);
  const float sh = beta[c] - mean[c] * sc;
  const _Float16* p = x + ((size_t)b * C + c) * Din * Din * Din;
  float res;
  if (pool) {
    res = -1e30f;
    for (int dz = 0; dz < 2; ++dz)
      for (int dy = 0; dy < 2; ++dy)
        for (int dx = 0; dx < 2; ++dx) {
          float v = (float)p[(size_t)(2 * z + dz) * Din * Din +
                             (2 * yy + dy) * Din + (2 * xx + dx)];
          res = fmaxf(res, fmaxf(v * sc + sh, 0.0f));
        }
  } else {
    const float v = (float)p[(size_t)z * Din * Din + yy * Din + xx];
    res = fmaxf(v * sc + sh, 0.0f);
  }
  y[idx] = (_Float16)res;
  if (y32) y32[idx] = res;
}

// ---------------------------------------------------------------------------
// 5) Small FC: y[b,o] = relu?(dot(x[b,:], w[o,:]) + bias[o])
// ---------------------------------------------------------------------------
__global__ __launch_bounds__(256) void fc_kernel(const float* __restrict__ x,
                                                 const float* __restrict__ w,
                                                 const float* __restrict__ bias,
                                                 float* __restrict__ y,
                                                 int K, int N, int relu) {
  const int idx = blockIdx.x * 256 + threadIdx.x;
  if (idx >= B_ * N) return;
  const int b = idx / N, o = idx % N;
  const float* xb = x + (size_t)b * K;
  const float* wo = w + (size_t)o * K;
  float s = bias[o];
  for (int k = 0; k < K; ++k) s = fmaf(xb[k], wo[k], s);
  if (relu) s = fmaxf(s, 0.0f);
  y[idx] = s;
}

// ---------------------------------------------------------------------------
// 6) Heads (tanh/sigmoid) + sample_lv_points, one block.
//    nodes: [8, S*(NPTS+1)=396, 3] f32, index s*33+i (i=32 -> center).
// ---------------------------------------------------------------------------
__global__ __launch_bounds__(256) void geometry(
    const float* __restrict__ h, const float* __restrict__ e1w,
    const float* __restrict__ e1b, const float* __restrict__ e2w,
    const float* __restrict__ e2b, const float* __restrict__ e3w,
    const float* __restrict__ e3b, float* __restrict__ nodes) {
  __shared__ float par1[B_][4], par2[B_], par3[B_][33];
  __shared__ float cx[B_][S_], cy[B_][S_], zc[B_][S_], rad[B_][S_];
  const int tid = threadIdx.x;
  for (int t = tid; t < B_ * 38; t += 256) {
    const int b = t / 38, o = t % 38;
    const float* xb = h + b * 200;
    const float* wrow;
    float bias;
    if (o < 4)       { wrow = e1w + o * 200;       bias = e1b[o]; }
    else if (o == 4) { wrow = e2w;                 bias = e2b[0]; }
    else             { wrow = e3w + (o - 5) * 200; bias = e3b[o - 5]; }
    float s = bias;
    for (int k = 0; k < 200; ++k) s = fmaf(xb[k], wrow[k], s);
    if (o < 4)       par1[b][o]     = tanhf(s);
    else if (o == 4) par2[b]        = sigm(s);
    else             par3[b][o - 5] = tanhf(s);
  }
  __syncthreads();
  if (tid < B_) {
    const int b = tid;
    const float cx0 = par1[b][0], cy0 = par1[b][1], z0 = par1[b][2];
    const float r0 = fabsf(par1[b][3]) + 0.05f;
    const float height = 2.0f * par2[b];
    float sdx = 0.0f, sdy = 0.0f, sdr = 0.0f;
    for (int s = 0; s < S_; ++s) {
      if (s > 0) {
        sdx += par3[b][s - 1];
        sdy += par3[b][11 + s - 1];
        sdr += fabsf(par3[b][22 + s - 1]);
      }
      const float frac = (float)s / (float)(S_ - 1);
      cx[b][s] = cx0 + 0.1f * sdx;
      cy[b][s] = cy0 + 0.1f * sdy;
      zc[b][s] = z0 + height * frac;
      float tt = fminf(fmaxf(1.0f - frac * frac, 0.0f), 1.0f);
      rad[b][s] = fmaxf(r0 * sqrtf(tt) - 0.1f * sdr, 0.02f);
    }
  }
  __syncthreads();
  for (int t = tid; t < B_ * S_ * 33; t += 256) {
    const int b = t / (S_ * 33);
    const int r = t % (S_ * 33);
    const int s = r / 33, i = r % 33;
    float px, py;
    const float pz = zc[b][s];
    if (i < NPTS_) {
      const float th = 6.28318530717958647692f * (float)i / (float)NPTS_;
      px = cx[b][s] + rad[b][s] * __cosf(th);
      py = cy[b][s] + rad[b][s] * __sinf(th);
    } else {
      px = cx[b][s];
      py = cy[b][s];
    }
    float* np_ = nodes + (size_t)t * 3;
    np_[0] = px; np_[1] = py; np_[2] = pz;
  }
}

// ---------------------------------------------------------------------------
// 7) Per-tetra precompute: v0 and rows (cross/det) so lam_i = dot(d, row_i).
//    Tetra indices generated analytically from build_tetras().
// ---------------------------------------------------------------------------
__global__ __launch_bounds__(256) void tet_precompute(
    const float* __restrict__ nodes, float* __restrict__ tets) {
  const int idx = blockIdx.x * 256 + threadIdx.x;
  if (idx >= B_ * NTET) return;
  const int b = idx / NTET, t = idx % NTET;
  const int m = t % 3;
  const int si = t / 3;
  const int s = si / NPTS_, i = si % NPTS_;
  const int j = (i + 1) % NPTS_;
  const int cen0 = s * 33 + 32, cen1 = (s + 1) * 33 + 32;
  const int ri = s * 33 + i, rj = s * 33 + j;
  const int ri1 = (s + 1) * 33 + i, rj1 = (s + 1) * 33 + j;
  int n0, n1, n2, n3;
  if (m == 0)      { n0 = cen0; n1 = ri;   n2 = rj;  n3 = rj1; }
  else if (m == 1) { n0 = cen0; n1 = ri;   n2 = ri1; n3 = rj1; }
  else             { n0 = cen0; n1 = cen1; n2 = ri1; n3 = rj1; }
  const float* nb = nodes + (size_t)b * (S_ * 33) * 3;
  const float v0x = nb[n0 * 3], v0y = nb[n0 * 3 + 1], v0z = nb[n0 * 3 + 2];
  const float e1x = nb[n1 * 3] - v0x, e1y = nb[n1 * 3 + 1] - v0y, e1z = nb[n1 * 3 + 2] - v0z;
  const float e2x = nb[n2 * 3] - v0x, e2y = nb[n2 * 3 + 1] - v0y, e2z = nb[n2 * 3 + 2] - v0z;
  const float e3x = nb[n3 * 3] - v0x, e3y = nb[n3 * 3 + 1] - v0y, e3z = nb[n3 * 3 + 2] - v0z;
  const float c0x = e2y * e3z - e2z * e3y, c0y = e2z * e3x - e2x * e3z, c0z = e2x * e3y - e2y * e3x;
  const float c1x = e3y * e1z - e3z * e1y, c1y = e3z * e1x - e3x * e1z, c1z = e3x * e1y - e3y * e1x;
  const float c2x = e1y * e2z - e1z * e2y, c2y = e1z * e2x - e1x * e2z, c2z = e1x * e2y - e1y * e2x;
  float det = e1x * c0x + e1y * c0y + e1z * c0z;
  det += (det >= 0.0f ? 1e-4f : -1e-4f);
  const float inv = 1.0f / det;
  float* o = tets + (size_t)idx * 12;
  o[0] = v0x;       o[1] = v0y;       o[2] = v0z;
  o[3] = c0x * inv; o[4] = c0y * inv; o[5] = c0z * inv;
  o[6] = c1x * inv; o[7] = c1y * inv; o[8] = c1z * inv;
  o[9] = c2x * inv; o[10] = c2y * inv; o[11] = c2z * inv;
}

// ---------------------------------------------------------------------------
// 8) Voxelize: each thread = one grid point, loop all tetras (LDS-staged).
// ---------------------------------------------------------------------------
__global__ __launch_bounds__(256) void voxelize_k(const float* __restrict__ tets,
                                                  float* __restrict__ out) {
  const int b = blockIdx.y;
  const int p = blockIdx.x * 256 + threadIdx.x;   // 0..32767
  const int ix = p >> 10, iy = (p >> 5) & 31, iz = p & 31;
  const float gx = ((float)ix + 0.5f) * (2.0f / 32.0f) - 1.0f;
  const float gy = ((float)iy + 0.5f) * (2.0f / 32.0f) - 1.0f;
  const float gz = ((float)iz + 0.5f) * (2.0f / 32.0f) - 1.0f;
  __shared__ float Ts[32 * 12];
  float occ = 0.0f;
  const float* tb = tets + (size_t)b * NTET * 12;
  for (int t0 = 0; t0 < NTET; t0 += 32) {
    __syncthreads();
    for (int e = threadIdx.x; e < 32 * 12; e += 256)
      Ts[e] = tb[(size_t)t0 * 12 + e];
    __syncthreads();
    for (int tl = 0; tl < 32; ++tl) {
      const float* T = Ts + tl * 12;
      const float dx = gx - T[0], dy = gy - T[1], dz = gz - T[2];
      const float l0 = dx * T[3] + dy * T[4] + dz * T[5];
      const float l1 = dx * T[6] + dy * T[7] + dz * T[8];
      const float l2 = dx * T[9] + dy * T[10] + dz * T[11];
      const float lr = 1.0f - (l0 + l1 + l2);
      const float sv = sigm(KSH * l0) * sigm(KSH * l1) * sigm(KSH * l2) * sigm(KSH * lr);
      occ = fmaxf(occ, sv);
    }
  }
  out[(size_t)b * 32768 + p] = occ;
}

// ---------------------------------------------------------------------------
// Host orchestration
// ---------------------------------------------------------------------------
extern "C" void kernel_launch(void* const* d_in, const int* in_sizes, int n_in,
                              void* d_out, int out_size, void* d_ws, size_t ws_size,
                              hipStream_t stream) {
  (void)in_sizes; (void)n_in; (void)out_size; (void)ws_size;
  // Input order per setup_inputs(): img, then params in insertion order.
  const float* img = (const float*)d_in[0];
  const float *enc_w[4], *enc_b[4], *enc_g[4], *enc_beta[4];
  for (int i = 0; i < 4; ++i) {
    enc_w[i]    = (const float*)d_in[1 + 4 * i];
    enc_b[i]    = (const float*)d_in[2 + 4 * i];
    enc_g[i]    = (const float*)d_in[3 + 4 * i];
    enc_beta[i] = (const float*)d_in[4 + 4 * i];
  }
  const float* reg_w0 = (const float*)d_in[17];
  const float* reg_b0 = (const float*)d_in[18];
  const float* reg_g0 = (const float*)d_in[19];
  const float* reg_be0 = (const float*)d_in[20];
  const float* reg_w1 = (const float*)d_in[21];
  const float* reg_b1 = (const float*)d_in[22];
  const float* reg_g1 = (const float*)d_in[23];
  const float* reg_be1 = (const float*)d_in[24];
  const float* fc1_w = (const float*)d_in[25];
  const float* fc1_b = (const float*)d_in[26];
  const float* fc2_w = (const float*)d_in[27];
  const float* fc2_b = (const float*)d_in[28];
  const float* end1_w = (const float*)d_in[29];
  const float* end1_b = (const float*)d_in[30];
  const float* end2_w = (const float*)d_in[31];
  const float* end2_b = (const float*)d_in[32];
  const float* end3_w = (const float*)d_in[33];
  const float* end3_b = (const float*)d_in[34];

  // Workspace layout (~52 MB total).
  char* cur = (char*)d_ws;
  auto alloc = [&cur](size_t bytes) -> void* {
    void* p = (void*)cur;
    cur += (bytes + 255) & ~(size_t)255;
    return p;
  };
  _Float16* nimg = (_Float16*)alloc((size_t)262144 * 2);        // 8x1x32^3
  _Float16* c0   = (_Float16*)alloc((size_t)16777216 * 2);      // 8x64x32^3
  _Float16* a0   = (_Float16*)alloc((size_t)2097152 * 2);       // 8x64x16^3
  _Float16* c1   = (_Float16*)alloc((size_t)4194304 * 2);       // 8x128x16^3
  _Float16* a1   = (_Float16*)alloc((size_t)524288 * 2);        // 8x128x8^3
  _Float16* c2   = (_Float16*)alloc((size_t)1048576 * 2);       // 8x256x8^3
  _Float16* a2   = (_Float16*)alloc((size_t)131072 * 2);        // 8x256x4^3
  _Float16* c3   = (_Float16*)alloc((size_t)262144 * 2);        // 8x512x4^3
  _Float16* a3   = (_Float16*)alloc((size_t)262144 * 2);        // 8x512x4^3
  _Float16* c4   = (_Float16*)alloc((size_t)131072 * 2);        // 8x256x4^3
  _Float16* a4   = (_Float16*)alloc((size_t)16384 * 2);         // 8x256x2^3
  _Float16* c5   = (_Float16*)alloc((size_t)8192 * 2);          // 8x128x2^3
  _Float16* a5   = (_Float16*)alloc((size_t)8192 * 2);          // 8x128x2^3 f16
  float* fcin  = (float*)alloc((size_t)8192 * 4);               // 8x1024
  float* h1    = (float*)alloc((size_t)3200 * 4);               // 8x400
  float* h2    = (float*)alloc((size_t)1600 * 4);               // 8x200
  float* nodes = (float*)alloc((size_t)9504 * 4);               // 8x396x3
  float* tets  = (float*)alloc((size_t)(B_ * NTET * 12) * 4);   // 8x1056x12
  float* meanb = (float*)alloc((size_t)512 * 4);
  float* varb  = (float*)alloc((size_t)512 * 4);

  float* mask = (float*)d_out;  // 8x1x32^3 f32

  // --- normalize ---
  norm_input<<<B_, 256, 0, stream>>>(img, nimg);

  // --- encoder ---  (N tiles of 128 columns, M tiles of 32 couts)
  // L0: 1 -> 64 @32, pool -> 16            N=262144
  conv3d_wmma<<<dim3(2048, 2), 256, 0, stream>>>(nimg, enc_w[0], enc_b[0], c0,
                                                 1, 64, 32, 5, 2);
  bn_stats<<<64, 256, 0, stream>>>(c0, meanb, varb, 64, 32768);
  bn_relu_pool<<<8192, 256, 0, stream>>>(c0, meanb, varb, enc_g[0], enc_beta[0],
                                         a0, nullptr, 64, 32, 1);
  // L1: 64 -> 128 @16, pool -> 8           N=32768
  conv3d_wmma<<<dim3(256, 4), 256, 0, stream>>>(a0, enc_w[1], enc_b[1], c1,
                                                64, 128, 16, 5, 2);
  bn_stats<<<128, 256, 0, stream>>>(c1, meanb, varb, 128, 4096);
  bn_relu_pool<<<2048, 256, 0, stream>>>(c1, meanb, varb, enc_g[1], enc_beta[1],
                                         a1, nullptr, 128, 16, 1);
  // L2: 128 -> 256 @8, pool -> 4           N=4096
  conv3d_wmma<<<dim3(32, 8), 256, 0, stream>>>(a1, enc_w[2], enc_b[2], c2,
                                               128, 256, 8, 5, 2);
  bn_stats<<<256, 256, 0, stream>>>(c2, meanb, varb, 256, 512);
  bn_relu_pool<<<512, 256, 0, stream>>>(c2, meanb, varb, enc_g[2], enc_beta[2],
                                        a2, nullptr, 256, 8, 1);
  // L3: 256 -> 512 @4, no pool             N=512
  conv3d_wmma<<<dim3(4, 16), 256, 0, stream>>>(a2, enc_w[3], enc_b[3], c3,
                                               256, 512, 4, 5, 2);
  bn_stats<<<512, 256, 0, stream>>>(c3, meanb, varb, 512, 64);
  bn_relu_pool<<<1024, 256, 0, stream>>>(c3, meanb, varb, enc_g[3], enc_beta[3],
                                         a3, nullptr, 512, 4, 0);
  // reg0: 512 -> 256 @4, pool -> 2         N=512
  conv3d_wmma<<<dim3(4, 8), 256, 0, stream>>>(a3, reg_w0, reg_b0, c4,
                                              512, 256, 4, 5, 2);
  bn_stats<<<256, 256, 0, stream>>>(c4, meanb, varb, 256, 64);
  bn_relu_pool<<<64, 256, 0, stream>>>(c4, meanb, varb, reg_g0, reg_be0,
                                       a4, nullptr, 256, 4, 1);
  // reg1: 256 -> 128 @2, 1x1x1, no pad     N=64 (partial N tile)
  conv3d_wmma<<<dim3(1, 4), 256, 0, stream>>>(a4, reg_w1, reg_b1, c5,
                                              256, 128, 2, 1, 0);
  bn_stats<<<128, 256, 0, stream>>>(c5, meanb, varb, 128, 8);
  bn_relu_pool<<<32, 256, 0, stream>>>(c5, meanb, varb, reg_g1, reg_be1,
                                       a5, fcin, 128, 2, 0);

  // --- FC head ---
  fc_kernel<<<(B_ * 400 + 255) / 256, 256, 0, stream>>>(fcin, fc1_w, fc1_b, h1,
                                                        1024, 400, 1);
  fc_kernel<<<(B_ * 200 + 255) / 256, 256, 0, stream>>>(h1, fc2_w, fc2_b, h2,
                                                        400, 200, 1);
  // --- geometry + voxelize ---
  geometry<<<1, 256, 0, stream>>>(h2, end1_w, end1_b, end2_w, end2_b,
                                  end3_w, end3_b, nodes);
  tet_precompute<<<(B_ * NTET + 255) / 256, 256, 0, stream>>>(nodes, tets);
  voxelize_k<<<dim3(128, B_), 256, 0, stream>>>(tets, mask);
}
